// VectorQuantizer_15487652069630
// MI455X (gfx1250) — matrix operand order repaired and checked by reading, not compiled, not run on hardware
//
#include <hip/hip_runtime.h>

typedef __attribute__((ext_vector_type(16))) _Float16 v16h;
typedef __attribute__((ext_vector_type(8)))  float    v8f;

#define B_  16
#define D_  128
#define T_  2048
#define K_  1024
#define N_  (B_*T_)          // 32768 rows
#define NT_ (N_/16)          // 2048 row tiles of 16
// workspace layout (bytes)
#define WS_LOSS   0
#define WS_EHI    64
#define WS_ELO    (64 + K_*D_*2)
#define WS_ENORM  (64 + 2*K_*D_*2)

// ---------------------------------------------------------------------------
// Kernel A: split embeddings into f16 hi/lo, precompute 0.5*|e|^2, zero loss.
// ---------------------------------------------------------------------------
__global__ void vq_prep(const float* __restrict__ emb,
                        float* __restrict__ ws_loss,
                        _Float16* __restrict__ ehi,
                        _Float16* __restrict__ elo,
                        float* __restrict__ enorm05) {
  int k = blockIdx.x * blockDim.x + threadIdx.x;
  if (k == 0) *ws_loss = 0.0f;
  if (k >= K_) return;
  const float* row = emb + (size_t)k * D_;
  float nrm = 0.0f;
  #pragma unroll 4
  for (int d = 0; d < D_; ++d) {
    float e = row[d];
    _Float16 h = (_Float16)e;                  // RNE
    _Float16 l = (_Float16)(e - (float)h);     // residual
    ehi[(size_t)k * D_ + d] = h;
    elo[(size_t)k * D_ + d] = l;
    nrm += e * e;
  }
  enorm05[k] = 0.5f * nrm;
}

// ---------------------------------------------------------------------------
// Kernel B: per 16-row tile: stage x via LDS, compensated-f16 WMMA scoring
// over all 1024 codes, argmax reduce, gather + write quantized, loss sum.
// ---------------------------------------------------------------------------
__launch_bounds__(128)
__global__ void vq_main(const float* __restrict__ x,
                        const float* __restrict__ emb,
                        const _Float16* __restrict__ ehi,
                        const _Float16* __restrict__ elo,
                        const float* __restrict__ enorm05,
                        float* __restrict__ out,
                        float* __restrict__ ws_loss) {
  __shared__ float sx[4][D_][20];   // padded: stride 20 dwords kills conflicts
  __shared__ int   sidx[4][16];

  const int w    = threadIdx.x >> 5;       // wave in block (0..3)
  const int lane = threadIdx.x & 31;
  const int m    = lane & 15;              // row (A) / column (B) within tile
  const int g    = lane >> 4;              // lane group (ISA A/B layout)
  const int tile = blockIdx.x * 4 + w;     // 0..2047
  const int b    = tile >> 7;              // batch index (T/16 = 128 tiles/b)
  const int t0   = (tile & 127) << 4;

  // ---- stage x tile [16 t x 128 d] into LDS, coalesced float4 ----
  {
    const int dl = lane >> 2;              // 0..7
    const int ts = (lane & 3) << 2;        // 0,4,8,12
    #pragma unroll
    for (int it = 0; it < 16; ++it) {
      int d = it * 8 + dl;
      float4 v = *(const float4*)(x + (size_t)b * D_ * T_ + (size_t)d * T_ + t0 + ts);
      *(float4*)&sx[w][d][ts] = v;
    }
  }
  __syncthreads();

  // ---- A tile -> f16 hi/lo in WMMA 16x32 layout (ISA 7.12.2) ----
  // element e of v16h: K = 32c + (e<8 ? 8g+e : 16+8g+(e-8))
  v16h ahi[4], alo[4];
  #pragma unroll
  for (int c = 0; c < 4; ++c) {
    #pragma unroll
    for (int e = 0; e < 16; ++e) {
      int d = c * 32 + ((e < 8) ? (g * 8 + e) : (16 + g * 8 + (e - 8)));
      float f = sx[w][d][m];
      _Float16 h = (_Float16)f;
      ahi[c][e] = h;
      alo[c][e] = (_Float16)(f - (float)h);
    }
  }

  // ---- scan codes: maximize s = x.e - 0.5|e|^2  (== argmin d2) ----
  float best[8]; int bidx[8];
  #pragma unroll
  for (int i = 0; i < 8; ++i) { best[i] = -3.4e38f; bidx[i] = 0; }

  for (int n0 = 0; n0 < K_; n0 += 16) {
    const int n = n0 + m;                  // this lane's column/code
    const _Float16* ph = ehi + (size_t)n * D_ + g * 8;
    const _Float16* pl = elo + (size_t)n * D_ + g * 8;

    // Issue ALL B-tile loads for this code tile up front (one load clause;
    // compiler inserts partial s_wait_loadcnt as tiles are consumed).
    union V16 { uint4 u[2]; v16h v; };
    V16 bh[4], bl[4];
    #pragma unroll
    for (int c = 0; c < 4; ++c) {
      bh[c].u[0] = *(const uint4*)(ph + c * 32);
      bh[c].u[1] = *(const uint4*)(ph + c * 32 + 16);
      bl[c].u[0] = *(const uint4*)(pl + c * 32);
      bl[c].u[1] = *(const uint4*)(pl + c * 32 + 16);
    }
    // Prefetch next code tile (+16 codes = +4KB); speculative, safe at end.
    __builtin_prefetch(ph + 16 * D_, 0, 0);   // -> global_prefetch_b8
    __builtin_prefetch(pl + 16 * D_, 0, 0);

    const float ev = enorm05[n];
    v8f acc;
    #pragma unroll
    for (int i = 0; i < 8; ++i) acc[i] = -ev;

    #pragma unroll
    for (int c = 0; c < 4; ++c) {
      // compensated product: hi*hi + lo*hi + hi*lo  (f32 accumulate)
      acc = __builtin_amdgcn_wmma_f32_16x16x32_f16(false, ahi[c], false, bh[c].v,
                                                   (short)0, acc, false, false);
      acc = __builtin_amdgcn_wmma_f32_16x16x32_f16(false, alo[c], false, bh[c].v,
                                                   (short)0, acc, false, false);
      acc = __builtin_amdgcn_wmma_f32_16x16x32_f16(false, ahi[c], false, bl[c].v,
                                                   (short)0, acc, false, false);
    }
    #pragma unroll
    for (int i = 0; i < 8; ++i) {
      if (acc[i] > best[i]) { best[i] = acc[i]; bidx[i] = n; }
    }
  }

  // ---- reduce across the 16 lanes sharing each row (xor 1,2,4,8) ----
  #pragma unroll
  for (int sh = 0; sh < 4; ++sh) {
    const int off = 1 << sh;
    #pragma unroll
    for (int i = 0; i < 8; ++i) {
      float os = __shfl_xor(best[i], off, 32);
      int   oi = __shfl_xor(bidx[i], off, 32);
      // higher score wins; on tie, lower code index (reference argmin rule)
      if (os > best[i] || (os == best[i] && oi < bidx[i])) {
        best[i] = os; bidx[i] = oi;
      }
    }
  }
  if (m == 0) {                            // lanes 0 (rows 0-7) and 16 (rows 8-15)
    #pragma unroll
    for (int i = 0; i < 8; ++i) sidx[w][g * 8 + i] = bidx[i];
  }
  __syncthreads();

  // ---- gather codes, write quantized [B,D,T], accumulate loss ----
  const int r    = m;                      // row in tile -> t = t0 + r
  const int idxr = sidx[w][r];
  const float* erow = emb + (size_t)idxr * D_;
  float lsum = 0.0f;
  #pragma unroll 8
  for (int j = 0; j < 64; ++j) {
    int d = 2 * j + g;
    float q  = erow[d];
    float xv = sx[w][d][r];
    float qq = xv + (q - xv);              // straight-through, reference rounding
    out[(size_t)b * D_ * T_ + (size_t)d * T_ + t0 + r] = qq;
    float diff = q - xv;
    lsum += diff * diff;
  }
  #pragma unroll
  for (int sh = 0; sh < 5; ++sh) lsum += __shfl_xor(lsum, 1 << sh, 32);
  if (lane == 0) atomicAdd(ws_loss, lsum);
}

// ---------------------------------------------------------------------------
// Kernel C: scale loss into the last output element.
// ---------------------------------------------------------------------------
__global__ void vq_finalize(const float* __restrict__ ws_loss,
                            float* __restrict__ out, int out_size) {
  if (blockIdx.x == 0 && threadIdx.x == 0) {
    out[out_size - 1] = 1.25f * (*ws_loss) * (1.0f / ((float)N_ * (float)D_));
  }
}

// ---------------------------------------------------------------------------
extern "C" void kernel_launch(void* const* d_in, const int* in_sizes, int n_in,
                              void* d_out, int out_size, void* d_ws, size_t ws_size,
                              hipStream_t stream) {
  const float* x   = (const float*)d_in[0];   // [16,1,128,2048] f32
  const float* emb = (const float*)d_in[1];   // [1024,128] f32
  float* out = (float*)d_out;                 // [16,128,2048] + loss scalar

  char* ws = (char*)d_ws;
  float*    wsl  = (float*)(ws + WS_LOSS);
  _Float16* ehi  = (_Float16*)(ws + WS_EHI);
  _Float16* elo  = (_Float16*)(ws + WS_ELO);
  float*    en05 = (float*)(ws + WS_ENORM);

  vq_prep<<<dim3(8), dim3(128), 0, stream>>>(emb, wsl, ehi, elo, en05);
  vq_main<<<dim3(NT_ / 4), dim3(128), 0, stream>>>(x, emb, ehi, elo, en05, out, wsl);
  vq_finalize<<<dim3(1), dim3(1), 0, stream>>>(wsl, out, out_size);
}
